// MultiHeadAttention_32650341384849
// MI455X (gfx1250) — matrix operand order repaired and checked by reading, not compiled, not run on hardware
//
#include <hip/hip_runtime.h>

// MI455X (gfx1250) fused multi-head attention, bf16 WMMA + async-to-LDS staging.
// prep_x:  x f32 -> x_bf bf16 row-major                      [one-shot]
// prep_w:  Wq/Wk/Wv/Wo f32 -> transposed bf16 [n][k]         [one-shot]
// qkv:     double-buffered async-LDS GEMM (k-step 64) -> Q/K/V bf16 [B,H,S,64]
// flash:   causal flash attention (async K staging) -> ctx bf16 [B*S,1024]
// oproj:   double-buffered async-LDS GEMM (+bo) -> d_out f32 [B,S,1024]

#define DM 1024
#define HH 16
#define DK 64
#define SQ 2048
#define BB 4

typedef __attribute__((ext_vector_type(16))) __bf16 bf16x16;
typedef __attribute__((ext_vector_type(8)))  float  f32x8;

union Frag {
  bf16x16 v;
  uint4   q[2];
};

__device__ __forceinline__ unsigned short f2bf(float f) {
  __bf16 h = (__bf16)f;                       // hardware cvt (RNE)
  return __builtin_bit_cast(unsigned short, h);
}
__device__ __forceinline__ unsigned pk2(float a, float b) {
  return (unsigned)f2bf(a) | ((unsigned)f2bf(b) << 16);
}
__device__ __forceinline__ f32x8 wmma_bf16(const Frag& a, const Frag& b, f32x8 c) {
  return __builtin_amdgcn_wmma_f32_16x16x32_bf16(false, a.v, false, b.v,
                                                 (short)0, c, false, false);
}

// LDS byte offset of a __shared__ object (low 32 bits of flat addr = LDS offset)
__device__ __forceinline__ unsigned lds_off(const void* p) {
  return (unsigned)(unsigned long long)p;
}
// 16B/lane async copy global -> LDS, tracked by ASYNCcnt.
__device__ __forceinline__ void async_b128(unsigned ldsoff, const void* g) {
  asm volatile("global_load_async_to_lds_b128 %0, %1, off"
               :: "v"(ldsoff), "v"((unsigned long long)g) : "memory");
}
__device__ __forceinline__ void wait_async0() {
  asm volatile("s_wait_asynccnt 0x0" ::: "memory");
}

// ---------------------------------------------------------------------------
// Prep 1: x f32 -> bf16, row-major.  8M elements, 8 per thread.
// ---------------------------------------------------------------------------
__global__ __launch_bounds__(256)
void prep_x(const float* __restrict__ x, unsigned short* __restrict__ xb) {
  size_t i = ((size_t)blockIdx.x * 256 + threadIdx.x) * 8;
  float4 a = *(const float4*)(x + i);
  float4 b = *(const float4*)(x + i + 4);
  uint4 o;
  o.x = pk2(a.x, a.y); o.y = pk2(a.z, a.w);
  o.z = pk2(b.x, b.y); o.w = pk2(b.z, b.w);
  *(uint4*)(xb + i) = o;
}

// ---------------------------------------------------------------------------
// Prep 2: weights -> transposed bf16.
//  y=0..2: Wq/Wk/Wv [H][DM][DK] -> wt[((y*H+h)*DK+n)*DM + d]
//  y=3   : Wo [DM][DM]          -> wt[3*H*DK*DM + n*DM + k]
// ---------------------------------------------------------------------------
__global__ __launch_bounds__(256)
void prep_w(const float* __restrict__ Wq, const float* __restrict__ Wk,
            const float* __restrict__ Wv, const float* __restrict__ Wo,
            unsigned short* __restrict__ wt) {
  const int y = blockIdx.y;
  const size_t i = (size_t)blockIdx.x * 256 + threadIdx.x;   // 0 .. 1M-1
  const size_t per = (size_t)HH * DK * DM;                   // 1,048,576
  if (y < 3) {
    const float* W = (y == 0 ? Wq : (y == 1 ? Wk : Wv));
    int d = (int)(i & (DM - 1));
    int rest = (int)(i >> 10);
    int n = rest & (DK - 1);
    int h = rest >> 6;
    wt[(size_t)y * per + i] = f2bf(W[((size_t)h * DM + d) * DK + n]);
  } else {
    int k = (int)(i & (DM - 1));
    int n = (int)(i >> 10);
    wt[3 * per + i] = f2bf(Wo[(size_t)k * DM + n]);
  }
}

// ---------------------------------------------------------------------------
// Shared GEMM body: C[128x64] += A[128xDM] * B^T (B stored [n][k] bf16).
// Double-buffered async-LDS staging, k-step 64, branch-free loop body.
// ---------------------------------------------------------------------------
__device__ __forceinline__ void gemm_body(
    const unsigned short* __restrict__ A,   // [.][DM] bf16, row mbase..
    const unsigned short* __restrict__ Bt,  // [64+][DM] bf16, row nbase..
    int mbase, int nbase,
    unsigned short* ldsA,                   // [2][128*64]
    unsigned short* ldsB,                   // [2][64*64]
    f32x8 acc[4]) {
  const int tid  = threadIdx.x;
  const int lane = tid & 31, w = tid >> 5;
  const int l15  = lane & 15, hi = lane >> 4;
  const int mi   = w >> 1, ni = w & 1;

  // staging coordinates (per k-step slab: A 128x64 = 16KB, B 64x64 = 8KB)
  const int arow = tid >> 1, achk = (tid & 1) * 32;   // A: 4 x b128 / thread
  const int brow = tid >> 2, bchk = (tid & 3) * 16;   // B: 2 x b128 / thread
  const unsigned short* aSrc = A + (size_t)(mbase + arow) * DM + achk;
  const unsigned short* bSrc = Bt + (size_t)(nbase + brow) * DM + bchk;

  { // prologue: slab 0
    unsigned ao = lds_off(ldsA + arow * 64 + achk);
    async_b128(ao, aSrc);           async_b128(ao + 16, aSrc + 8);
    async_b128(ao + 32, aSrc + 16); async_b128(ao + 48, aSrc + 24);
    unsigned bo_ = lds_off(ldsB + brow * 64 + bchk);
    async_b128(bo_, bSrc);          async_b128(bo_ + 16, bSrc + 8);
  }

  const int N = DM / 64;                    // 16 k-steps
  for (int i = 0; i < N; ++i) {
    const int cur = i & 1;
    wait_async0();
    __syncthreads();
    { // issue next slab into other buffer; last iter re-copies (dead buffer)
      const int nx = (i + 1 < N) ? (i + 1) : i;
      const unsigned short* as = aSrc + nx * 64;
      const unsigned short* bs = bSrc + nx * 64;
      unsigned ao = lds_off(ldsA + (1 - cur) * (128 * 64) + arow * 64 + achk);
      async_b128(ao, as);           async_b128(ao + 16, as + 8);
      async_b128(ao + 32, as + 16); async_b128(ao + 48, as + 24);
      unsigned bo_ = lds_off(ldsB + (1 - cur) * (64 * 64) + brow * 64 + bchk);
      async_b128(bo_, bs);          async_b128(bo_ + 16, bs + 8);
    }
    const unsigned short* sa = ldsA + cur * (128 * 64);
    const unsigned short* sb = ldsB + cur * (64 * 64);
#pragma unroll
    for (int kq = 0; kq < 2; ++kq) {        // two 32-deep sub-steps
      Frag fa[2], fb[2];
#pragma unroll
      for (int m2 = 0; m2 < 2; ++m2) {
        const unsigned short* ap =
            sa + (mi * 32 + m2 * 16 + l15) * 64 + kq * 32 + hi * 8;
        fa[m2].q[0] = *(const uint4*)ap;
        fa[m2].q[1] = *(const uint4*)(ap + 16);
      }
#pragma unroll
      for (int n2 = 0; n2 < 2; ++n2) {
        const unsigned short* bp =
            sb + (ni * 32 + n2 * 16 + l15) * 64 + kq * 32 + hi * 8;
        fb[n2].q[0] = *(const uint4*)bp;
        fb[n2].q[1] = *(const uint4*)(bp + 16);
      }
#pragma unroll
      for (int m2 = 0; m2 < 2; ++m2)
#pragma unroll
        for (int n2 = 0; n2 < 2; ++n2)
          acc[m2 * 2 + n2] = wmma_bf16(fa[m2], fb[n2], acc[m2 * 2 + n2]);
    }
  }
}

// ---------------------------------------------------------------------------
// Kernel 1: QKV projection.  grid = (64 M-tiles, 3*H).  8 waves.
// ---------------------------------------------------------------------------
__global__ __launch_bounds__(256)
void qkv_kernel(const unsigned short* __restrict__ xb,
                const unsigned short* __restrict__ wt,
                const float* __restrict__ bq, const float* __restrict__ bk,
                const float* __restrict__ bv,
                unsigned short* __restrict__ q_ws,
                unsigned short* __restrict__ k_ws,
                unsigned short* __restrict__ v_ws) {
  __shared__ unsigned short ldsA[2][128 * 64];
  __shared__ unsigned short ldsB[2][64 * 64];

  const int tid  = threadIdx.x;
  const int lane = tid & 31, w = tid >> 5;
  const int l15  = lane & 15, hi = lane >> 4;
  const int mi   = w >> 1, ni = w & 1;
  const int t    = blockIdx.y / HH, h = blockIdx.y % HH;
  const int mbase = blockIdx.x * 128;

  const unsigned short* Wt = wt + (size_t)(t * HH + h) * DK * DM; // [64][1024]
  const float* bias = (t == 0 ? bq : (t == 1 ? bk : bv)) + h * DK;
  unsigned short* out = (t == 0 ? q_ws : (t == 1 ? k_ws : v_ws));

  f32x8 acc[4] = {};
  gemm_body(xb, Wt, mbase, 0, &ldsA[0][0], &ldsB[0][0], acc);

#pragma unroll
  for (int m2 = 0; m2 < 2; ++m2)
#pragma unroll
    for (int n2 = 0; n2 < 2; ++n2) {
      int col = ni * 32 + n2 * 16 + l15;
      float bb = bias[col];
#pragma unroll
      for (int r = 0; r < 8; ++r) {
        int m = mbase + mi * 32 + m2 * 16 + r + 8 * hi;  // row in [0, 8192)
        int bidx = m >> 11, s = m & (SQ - 1);
        out[(((size_t)(bidx * HH + h)) * SQ + s) * DK + col] =
            f2bf(acc[m2 * 2 + n2][r] + bb);
      }
    }
}

// ---------------------------------------------------------------------------
// Kernel 2: causal flash attention.  grid = (S/128, H, B).  8 waves, each
// owns 16 query rows; streams 32-key blocks; async K staging; double-buffered.
// ---------------------------------------------------------------------------
__global__ __launch_bounds__(256)
void flash_kernel(const unsigned short* __restrict__ q_ws,
                  const unsigned short* __restrict__ k_ws,
                  const unsigned short* __restrict__ v_ws,
                  unsigned short* __restrict__ ctx) {
  __shared__ unsigned short ldsK[2][32 * 64];  // K block row-major [key][d]
  __shared__ unsigned short ldsV[2][64 * 32];  // V block transposed [d][key]
  __shared__ unsigned short ldsP[8][16 * 32];  // per-wave P tile [row][key]

  const int tid  = threadIdx.x;
  const int lane = tid & 31, w = tid >> 5;
  const int l15  = lane & 15, hi = lane >> 4;
  const int wgq  = blockIdx.x, h = blockIdx.y, b = blockIdx.z;
  const size_t bh = (size_t)(b * HH + h) * SQ * DK;
  const unsigned short* Q = q_ws + bh;
  const unsigned short* K = k_ws + bh;
  const unsigned short* V = v_ws + bh;
  const int m0 = wgq * 128 + w * 16;           // this wave's query base

  Frag qa[2];                                  // Q fragments live in regs
  {
    int row = m0 + l15;
#pragma unroll
    for (int ks = 0; ks < 2; ++ks) {
      const unsigned short* qp = Q + (size_t)row * DK + ks * 32 + hi * 8;
      qa[ks].q[0] = *(const uint4*)qp;
      qa[ks].q[1] = *(const uint4*)(qp + 16);
    }
  }

  f32x8 o[4] = {};
  float rm[8], rl[8];
#pragma unroll
  for (int r = 0; r < 8; ++r) { rm[r] = -1e30f; rl[r] = 0.f; }

  const float scale = 0.125f;                  // 1/sqrt(64)
  const int nblk = (wgq * 128 + 128) / 32;
  const int stgKey = tid >> 3, stgD = (tid & 7) * 8;  // 16B per thread

  // prologue: stage block 0 (K async, V load+transpose-store)
  async_b128(lds_off(&ldsK[0][stgKey * 64 + stgD]),
             K + (size_t)stgKey * DK + stgD);
  {
    uint4 vv = *(const uint4*)(V + (size_t)stgKey * DK + stgD);
    const unsigned short* pv = (const unsigned short*)&vv;
#pragma unroll
    for (int j = 0; j < 8; ++j)
      ldsV[0][(stgD + j) * 32 + stgKey] = pv[j];
  }

  for (int i = 0; i < nblk; ++i) {
    const int kb = i * 32, cur = i & 1;
    wait_async0();
    __syncthreads();
    { // stage next block (clamped on last iter -> dead-buffer re-copy)
      const int kn = (i + 1 < nblk) ? kb + 32 : kb;
      async_b128(lds_off(&ldsK[1 - cur][stgKey * 64 + stgD]),
                 K + (size_t)(kn + stgKey) * DK + stgD);
      uint4 vv = *(const uint4*)(V + (size_t)(kn + stgKey) * DK + stgD);
      const unsigned short* pv = (const unsigned short*)&vv;
#pragma unroll
      for (int j = 0; j < 8; ++j)
        ldsV[1 - cur][(stgD + j) * 32 + stgKey] = pv[j];
    }

    if (kb <= m0 + 15) {                       // causal: block intersects range
      const unsigned short* kbuf = &ldsK[cur][0];
      const unsigned short* vbuf = &ldsV[cur][0];

      // ---- scores: Q (16x64) @ K^T -> two 16x16 tiles ----
      f32x8 s0 = {}, s1 = {};
#pragma unroll
      for (int ks = 0; ks < 2; ++ks) {
        Frag fk0, fk1;
        const unsigned short* kp0 = kbuf + (l15) * DK + ks * 32 + hi * 8;
        fk0.q[0] = *(const uint4*)kp0;
        fk0.q[1] = *(const uint4*)(kp0 + 16);
        const unsigned short* kp1 = kbuf + (16 + l15) * DK + ks * 32 + hi * 8;
        fk1.q[0] = *(const uint4*)kp1;
        fk1.q[1] = *(const uint4*)(kp1 + 16);
        s0 = wmma_bf16(qa[ks], fk0, s0);
        s1 = wmma_bf16(qa[ks], fk1, s1);
      }

      // ---- online softmax (row stats replicated across 16-lane halves) ----
      unsigned short* P = &ldsP[w][0];
#pragma unroll
      for (int r = 0; r < 8; ++r) {
        int qg = m0 + r + 8 * hi;
        float a0 = (kb + l15      <= qg) ? s0[r] * scale : -1e30f;
        float a1 = (kb + 16 + l15 <= qg) ? s1[r] * scale : -1e30f;
        float bm = fmaxf(a0, a1);
#pragma unroll
        for (int msk = 1; msk <= 8; msk <<= 1)
          bm = fmaxf(bm, __shfl_xor(bm, msk, 32));
        float mn = fmaxf(rm[r], bm);
        float sc = __expf(rm[r] - mn);
        float p0 = __expf(a0 - mn);
        float p1 = __expf(a1 - mn);
        float rs = p0 + p1;
#pragma unroll
        for (int msk = 1; msk <= 8; msk <<= 1)
          rs += __shfl_xor(rs, msk, 32);
        rl[r] = rl[r] * sc + rs;
        rm[r] = mn;
        int rowp = (r + 8 * hi) * 32;          // C-layout -> row-major P in LDS
        P[rowp + l15]      = f2bf(p0);
        P[rowp + 16 + l15] = f2bf(p1);
        o[0][r] *= sc; o[1][r] *= sc; o[2][r] *= sc; o[3][r] *= sc;
      }

      // ---- O += P (16x32) @ V (32x64) ----
      Frag pa;
      const unsigned short* pp = P + l15 * 32 + hi * 8;
      pa.q[0] = *(const uint4*)pp;
      pa.q[1] = *(const uint4*)(pp + 16);
#pragma unroll
      for (int j = 0; j < 4; ++j) {
        Frag vb;
        const unsigned short* vp = vbuf + (j * 16 + l15) * 32 + hi * 8;
        vb.q[0] = *(const uint4*)vp;
        vb.q[1] = *(const uint4*)(vp + 16);
        o[j] = wmma_bf16(pa, vb, o[j]);
      }
    }
  }

  // normalize and emit ctx (bf16) in [B*S, H*64] layout for the out-proj GEMM
  float rinv[8];
#pragma unroll
  for (int r = 0; r < 8; ++r) rinv[r] = 1.0f / rl[r];
#pragma unroll
  for (int j = 0; j < 4; ++j) {
    int col = h * DK + j * 16 + l15;
#pragma unroll
    for (int r = 0; r < 8; ++r) {
      int s = m0 + r + 8 * hi;
      ctx[((size_t)(b * SQ + s)) * DM + col] = f2bf(o[j][r] * rinv[r]);
    }
  }
}

// ---------------------------------------------------------------------------
// Kernel 3: out = ctx (bf16) @ Wo_t (+bo), f32 output.  grid = (64, 16).
// ---------------------------------------------------------------------------
__global__ __launch_bounds__(256)
void oproj_kernel(const unsigned short* __restrict__ ctx,
                  const unsigned short* __restrict__ wo_t,  // [n][k] bf16
                  const float* __restrict__ bo,
                  float* __restrict__ out) {
  __shared__ unsigned short ldsA[2][128 * 64];
  __shared__ unsigned short ldsB[2][64 * 64];

  const int tid  = threadIdx.x;
  const int lane = tid & 31, w = tid >> 5;
  const int l15  = lane & 15, hi = lane >> 4;
  const int mi   = w >> 1, ni = w & 1;
  const int mbase = blockIdx.x * 128;
  const int nbase = blockIdx.y * 64;

  f32x8 acc[4] = {};
  gemm_body(ctx, wo_t, mbase, nbase, &ldsA[0][0], &ldsB[0][0], acc);

#pragma unroll
  for (int m2 = 0; m2 < 2; ++m2)
#pragma unroll
    for (int n2 = 0; n2 < 2; ++n2) {
      int col = nbase + ni * 32 + n2 * 16 + l15;
      float bb = bo[col];
#pragma unroll
      for (int r = 0; r < 8; ++r) {
        int m = mbase + mi * 32 + m2 * 16 + r + 8 * hi;
        out[(size_t)m * DM + col] = acc[m2 * 2 + n2][r] + bb;
      }
    }
}

// ---------------------------------------------------------------------------
extern "C" void kernel_launch(void* const* d_in, const int* in_sizes, int n_in,
                              void* d_out, int out_size, void* d_ws, size_t ws_size,
                              hipStream_t stream) {
  const float* x  = (const float*)d_in[0];
  const float* Wq = (const float*)d_in[1];
  const float* bq = (const float*)d_in[2];
  const float* Wk = (const float*)d_in[3];
  const float* bk = (const float*)d_in[4];
  const float* Wv = (const float*)d_in[5];
  const float* bv = (const float*)d_in[6];
  const float* Wo = (const float*)d_in[7];
  const float* bo = (const float*)d_in[8];
  float* out = (float*)d_out;

  // workspace (bf16 elements):
  //  x_bf 8M | wt 4M (3M qkv-transposed + 1M wo-transposed) | Q,K,V 8M each | ctx 8M
  const size_t nX  = (size_t)BB * SQ * DM;       // 8M
  const size_t nWt = 4ull * HH * DK * DM;        // 4M
  const size_t per = (size_t)BB * HH * SQ * DK;  // 8M
  unsigned short* x_bf = (unsigned short*)d_ws;
  unsigned short* wt   = x_bf + nX;
  unsigned short* q_ws = wt + nWt;
  unsigned short* k_ws = q_ws + per;
  unsigned short* v_ws = k_ws + per;
  unsigned short* ctx  = v_ws + per;
  unsigned short* wo_t = wt + 3ull * HH * DK * DM;

  prep_x<<<dim3((unsigned)(nX / (256 * 8))), 256, 0, stream>>>(x, x_bf);
  prep_w<<<dim3(4096, 4), 256, 0, stream>>>(Wq, Wk, Wv, Wo, wt);
  qkv_kernel<<<dim3(64, 3 * HH), 256, 0, stream>>>(x_bf, wt, bq, bk, bv,
                                                   q_ws, k_ws, v_ws);
  flash_kernel<<<dim3(SQ / 128, HH, BB), 256, 0, stream>>>(q_ws, k_ws, v_ws, ctx);
  oproj_kernel<<<dim3(64, HH), 256, 0, stream>>>(ctx, wo_t, bo, out);
}
